// QuantisedLinear_7962869367159
// MI455X (gfx1250) — matrix operand order repaired and checked by reading, not compiled, not run on hardware
//
#include <hip/hip_runtime.h>
#include <hip/hip_bf16.h>

// QuantisedLinear: out[8,28672] = x[8,8192] . dequant(W)^T
// Memory-bound: streams ~470MB of int32-packed weights once (floor ~20us @23.3TB/s).
// v1 -> v2: pre-pass converts x to zero-padded f16 [16][8192] in d_ws, so the
// hot loop's A fragment is two 16B f16 loads (no cvt, no EXEC divergence);
// added next-block global_prefetch for the NT weight stream.

#define O_FEATS 28672
#define I_FEATS 8192
#define BATCH   8

typedef __attribute__((ext_vector_type(16))) _Float16 v16h;
typedef __attribute__((ext_vector_type(8)))  _Float16 v8h;
typedef __attribute__((ext_vector_type(8)))  float    v8f;
typedef __attribute__((ext_vector_type(4)))  int      v4i;
typedef __attribute__((ext_vector_type(4)))  float    v4f;

// ---------- pre-pass: x[8,8192] f32 -> xh[16,8192] f16, rows 8..15 = 0 ----------
__global__ __launch_bounds__(256)
void x_to_f16_padded(const float* __restrict__ x, _Float16* __restrict__ xh)
{
    const int idx = blockIdx.x * 256 + threadIdx.x;   // 16*8192 = 131072 threads
    const int row = idx >> 13;
    const int col = idx & (I_FEATS - 1);
    const float v = (row < BATCH) ? x[row * I_FEATS + col] : 0.0f;
    xh[idx] = (_Float16)v;
}

// ---------- main kernel (f16 staged A) ----------
__global__ __launch_bounds__(256)
void qlin_wmma_f16(const _Float16* __restrict__ xh,   // [16, 8192] zero-padded
                   const int*      __restrict__ wq,   // [28672, 4096]
                   const float*    __restrict__ scale,// [28672, 64]
                   const float*    __restrict__ lut,  // [256, 2]
                   float*          __restrict__ out)  // [8, 28672]
{
    __shared__ float2 lut_s[256];
    const int tid = threadIdx.x;
    lut_s[tid] = ((const float2*)lut)[tid];
    __syncthreads();

    const int wave = blockIdx.x * 8 + (tid >> 5);     // 1792 waves
    const int lane = tid & 31;
    const int o0   = wave << 4;
    const int n    = lane & 15;                       // output col / A row M
    const int hi   = lane >> 4;
    const int o    = o0 + n;

    const int*      wrow = wq + (size_t)o * (I_FEATS / 2);
    const _Float16* xrow = xh + (size_t)n * I_FEATS;
    const float*    srow = scale + (size_t)o * (I_FEATS / 128);

    v8f acc = {};

    for (int ib = 0; ib < I_FEATS / 128; ++ib) {
        const float s  = srow[ib];
        const int kblk = ib << 7;

        // Warm L2 for the next 128-input weight block while this one computes.
        if (ib + 1 < I_FEATS / 128)
            __builtin_prefetch(wrow + ((kblk + 128 + (hi << 4)) >> 1), 0, 0);

        #pragma unroll
        for (int cc = 0; cc < 4; ++cc) {              // 4 x K=32 WMMA per scale block
            const int k0 = kblk + (cc << 5);

            // ---- B fragment (32x16 f16): lanes 0-15 K=k0..+15, lanes 16-31 K=k0+16..+31.
            const int kb = k0 + (hi << 4);
            const v4i* wp = (const v4i*)(wrow + (kb >> 1));
            v4i w0 = __builtin_nontemporal_load(wp);          // NT b128 stream
            v4i w1 = __builtin_nontemporal_load(wp + 1);

            v16h bfrag;
            #pragma unroll
            for (int j = 0; j < 8; ++j) {
                const int bv   = ((j < 4) ? w0[j] : w1[j - 4]) & 0xFF;
                const float2 v = lut_s[bv];                    // ds_load_b64 gather
                bfrag[2 * j]     = (_Float16)(v.x * s);        // fold scale pre-cvt
                bfrag[2 * j + 1] = (_Float16)(v.y * s);
            }

            // ---- A fragment (16x32 f16, contiguous k order, same as B): two aligned 16B loads, no cvt.
            // lanes 0-15 (M=lane):   halves 0-7 = K k0+0..7,   8-15 = K k0+8..15
            // lanes 16-31(M=lane-16):halves 0-7 = K k0+16..23, 8-15 = K k0+24..31
            v8h a0 = *(const v8h*)(xrow + k0 + (hi << 4));
            v8h a1 = *(const v8h*)(xrow + k0 + 8 + (hi << 4));
            v16h afrag;
            #pragma unroll
            for (int j = 0; j < 8; ++j) {
                afrag[j]     = a0[j];
                afrag[8 + j] = a1[j];
            }

            acc = __builtin_amdgcn_wmma_f32_16x16x32_f16(
                      false, afrag, false, bfrag, (short)0, acc, false, false);
        }
    }

    // ---- D layout: VGPR r, lanes 0-15 -> (M=r, N=lane). Rows 0-7 are the batch.
    if (lane < 16) {
        float* op = out + o0 + lane;
        #pragma unroll
        for (int r = 0; r < BATCH; ++r)
            op[(size_t)r * O_FEATS] = acc[r];
    }
}

// ---------- fallback (no workspace): in-loop f32 x conversion ----------
__global__ __launch_bounds__(256)
void qlin_wmma_f16_f32x(const float* __restrict__ x,
                        const int*   __restrict__ wq,
                        const float* __restrict__ scale,
                        const float* __restrict__ lut,
                        float*       __restrict__ out)
{
    __shared__ float2 lut_s[256];
    const int tid = threadIdx.x;
    lut_s[tid] = ((const float2*)lut)[tid];
    __syncthreads();

    const int wave = blockIdx.x * 8 + (tid >> 5);
    const int lane = tid & 31;
    const int o0   = wave << 4;
    const int n    = lane & 15;
    const int hi   = lane >> 4;
    const int o    = o0 + n;

    const int*   wrow = wq + (size_t)o * (I_FEATS / 2);
    const float* xrow = x  + (size_t)n * I_FEATS;
    const float* srow = scale + (size_t)o * (I_FEATS / 128);
    const bool mvalid = (n < BATCH);

    v8f acc = {};
    for (int ib = 0; ib < I_FEATS / 128; ++ib) {
        const float s  = srow[ib];
        const int kblk = ib << 7;
        #pragma unroll
        for (int cc = 0; cc < 4; ++cc) {
            const int k0 = kblk + (cc << 5);
            const int kb = k0 + (hi << 4);
            const v4i* wp = (const v4i*)(wrow + (kb >> 1));
            v4i w0 = __builtin_nontemporal_load(wp);
            v4i w1 = __builtin_nontemporal_load(wp + 1);
            v16h bfrag;
            #pragma unroll
            for (int j = 0; j < 8; ++j) {
                const int bv   = ((j < 4) ? w0[j] : w1[j - 4]) & 0xFF;
                const float2 v = lut_s[bv];
                bfrag[2 * j]     = (_Float16)(v.x * s);
                bfrag[2 * j + 1] = (_Float16)(v.y * s);
            }
            v16h afrag = {};
            if (mvalid) {
                const v4f* xp = (const v4f*)(xrow + k0 + (hi << 4));
                v4f x0 = xp[0];
                v4f x1 = xp[1];
                const v4f* xq = (const v4f*)(xrow + k0 + 8 + (hi << 4));
                v4f x2 = xq[0];
                v4f x3 = xq[1];
                #pragma unroll
                for (int j = 0; j < 4; ++j) {
                    afrag[j]      = (_Float16)x0[j];
                    afrag[4 + j]  = (_Float16)x1[j];
                    afrag[8 + j]  = (_Float16)x2[j];
                    afrag[12 + j] = (_Float16)x3[j];
                }
            }
            acc = __builtin_amdgcn_wmma_f32_16x16x32_f16(
                      false, afrag, false, bfrag, (short)0, acc, false, false);
        }
    }
    if (lane < 16) {
        float* op = out + o0 + lane;
        #pragma unroll
        for (int r = 0; r < BATCH; ++r)
            op[(size_t)r * O_FEATS] = acc[r];
    }
}

extern "C" void kernel_launch(void* const* d_in, const int* in_sizes, int n_in,
                              void* d_out, int out_size, void* d_ws, size_t ws_size,
                              hipStream_t stream) {
    const float* x     = (const float*)d_in[0];
    const int*   wq    = (const int*)  d_in[1];
    const float* scale = (const float*)d_in[2];
    const float* lut   = (const float*)d_in[3];
    float*       out   = (float*)d_out;

    const size_t xh_bytes = (size_t)16 * I_FEATS * sizeof(_Float16);  // 256 KB

    dim3 grid(O_FEATS / 16 / 8);   // 224 blocks x 8 waves = 1792 tiles
    dim3 block(256);

    if (d_ws != nullptr && ws_size >= xh_bytes) {
        _Float16* xh = (_Float16*)d_ws;
        x_to_f16_padded<<<dim3((16 * I_FEATS) / 256), block, 0, stream>>>(x, xh);
        qlin_wmma_f16<<<grid, block, 0, stream>>>(xh, wq, scale, lut, out);
    } else {
        qlin_wmma_f16_f32x<<<grid, block, 0, stream>>>(x, wq, scale, lut, out);
    }
}
